// PosBiasedMultiHeadedAttention_17214228922450
// MI455X (gfx1250) — matrix operand order repaired and checked by reading, not compiled, not run on hardware
//
#include <hip/hip_runtime.h>
#include <stdint.h>

// ---------------- types ----------------
typedef __attribute__((ext_vector_type(16))) __bf16          v16bf;
typedef __attribute__((ext_vector_type(16))) unsigned short  ushort16;
typedef __attribute__((ext_vector_type(8)))  float           v8f;
typedef int v4i __attribute__((vector_size(4 * sizeof(int))));
typedef __attribute__((address_space(1))) v4i* gv4i_p;   // global int4*
typedef __attribute__((address_space(3))) v4i* lv4i_p;   // LDS int4*

#define D_MODEL   1024
#define H_HEADS   16
#define D_K       64
#define SEQ       512
#define BATCH     2
#define N_EMB     40001     // 2*1000*20 + 1
#define EMB_DIM   256

#if defined(__has_builtin)
#  if __has_builtin(__builtin_amdgcn_global_load_async_to_lds_b128)
#    define USE_ASYNC_BUILTIN 1
#  endif
#endif

__device__ __forceinline__ unsigned short to_bf16_u16(float f) {
    unsigned int u = __builtin_bit_cast(unsigned int, f);
    unsigned int r = u + 0x7FFFu + ((u >> 16) & 1u);   // round-to-nearest-even
    return (unsigned short)(r >> 16);
}

// async copy of 16B global -> LDS (per-lane addresses), tracked by ASYNCcnt
__device__ __forceinline__ void stage_async16(const unsigned short* g, unsigned short* l) {
#ifdef USE_ASYNC_BUILTIN
    __builtin_amdgcn_global_load_async_to_lds_b128(
        (gv4i_p)(void*)g, (lv4i_p)(void*)l, 0, 0);
#else
    unsigned int la = (unsigned int)(unsigned long long)
        (__attribute__((address_space(3))) unsigned short*)l;
    asm volatile("global_load_async_to_lds_b128 %0, %1, off"
                 :: "v"(la), "v"((unsigned long long)g) : "memory");
#endif
}
#define WAIT_ASYNC_LE1() asm volatile("s_wait_asynccnt 1" ::: "memory")
#define WAIT_ASYNC_LE0() asm volatile("s_wait_asynccnt 0" ::: "memory")

// ---------------- elementwise conversions ----------------
__global__ void k_f32_to_bf16(const float* __restrict__ in,
                              unsigned short* __restrict__ out, int n) {
    int i = blockIdx.x * blockDim.x + threadIdx.x;
    if (i < n) out[i] = to_bf16_u16(in[i]);
}

// wt[c][r] = w[r][c]   (store weights transposed so B-fragments load contiguously)
__global__ void k_transpose_bf16(const float* __restrict__ w,
                                 unsigned short* __restrict__ wt,
                                 int rows, int cols) {
    int c = blockIdx.x * blockDim.x + threadIdx.x;
    int r = blockIdx.y;
    if (c < cols) wt[(size_t)c * rows + r] = to_bf16_u16(w[(size_t)r * cols + c]);
}

// proj[r][h] = dot(table[r, h*16 : h*16+16], Wb)   -- collapses pos_emb @ Wb
__global__ void k_table_proj(const float* __restrict__ table,
                             const float* __restrict__ Wb,
                             float* __restrict__ proj, int nrows) {
    int t = blockIdx.x * blockDim.x + threadIdx.x;
    int r = t >> 4, h = t & 15;
    if (r < nrows) {
        const float* row = table + (size_t)r * EMB_DIM + h * 16;
        float s = 0.f;
#pragma unroll
        for (int d = 0; d < 16; ++d) s += row[d] * Wb[d];
        proj[t] = s;
    }
}

// bias[b,h,q,k] = sigmoid(px[ix][h] + py[iy][h] + bb), ix from query_x[b,k]-key_x[b,q]
__global__ void k_bias(const float* __restrict__ qx, const float* __restrict__ qy,
                       const float* __restrict__ kx, const float* __restrict__ ky,
                       const float* __restrict__ px, const float* __restrict__ py,
                       const float* __restrict__ bbp, float* __restrict__ bias) {
    int k = blockIdx.x * blockDim.x + threadIdx.x;   // key col
    int q = blockIdx.y;                               // query row
    int b = blockIdx.z;
    float bb = bbp[0];
    float dx = qx[b * SEQ + k] - kx[b * SEQ + q];
    float dy = qy[b * SEQ + k] - ky[b * SEQ + q];
    int ix = (int)rintf((fminf(fmaxf(dx, -1000.f), 1000.f) + 1000.f) * 20.f);
    int iy = (int)rintf((fminf(fmaxf(dy, -1000.f), 1000.f) + 1000.f) * 20.f);
    const float* pxr = px + (size_t)ix * 16;
    const float* pyr = py + (size_t)iy * 16;
    size_t base = (((size_t)b * H_HEADS) * SEQ + q) * SEQ + k;
#pragma unroll
    for (int h = 0; h < H_HEADS; ++h) {
        float z = pxr[h] + pyr[h] + bb;
        bias[base + (size_t)h * SEQ * SEQ] = 1.f / (1.f + __expf(-z));
    }
}

// ---------------- bf16 WMMA GEMM:  Y(1024x1024) = X @ W + bias ----------------
// Block = 128x64 output panel, 8 waves; wave w computes rows [w*16, w*16+16) x 64 cols.
// B slab (64 cols x 32 k, 4KB) staged into LDS with async-to-LDS, double-buffered;
// all 8 waves share it.  A fragments loaded directly (rows are wave-private).
// mode 0: out bf16 head-major (b,h,n,d_k), scaled (Q with 1/sqrt(dk), K)
// mode 1: out bf16 head-major transposed (b,h,d_k,n)  (V)
// mode 2: out f32 flat (final output projection)
__global__ __launch_bounds__(256)
void k_gemm(const unsigned short* __restrict__ X,
            const unsigned short* __restrict__ Wt,
            const float* __restrict__ bias,
            float scale, int mode,
            unsigned short* __restrict__ out_bf,
            float* __restrict__ out_f32) {
    __shared__ unsigned short bsh[2][64 * 32];   // [buf][n*32 + kk]
    int tid = threadIdx.x;
    int wid = tid >> 5, lane = tid & 31;
    int rb = blockIdx.x >> 4, cb = blockIdx.x & 15;   // 8 row panels x 16 col panels
    int m = lane & 15, half = lane >> 4;

    // staging role of this thread: one 16B chunk of the 4KB slab
    int sn = tid >> 2, sch = tid & 3;                 // slab row 0..63, chunk 0..3
    const unsigned short* gsrc = Wt + (size_t)(cb * 64 + sn) * D_MODEL + sch * 8;
    int lofs = sn * 32 + sch * 8;

    const unsigned short* xrow = X + (size_t)(rb * 128 + wid * 16 + m) * D_MODEL;

    v8f acc[4];
#pragma unroll
    for (int t = 0; t < 4; ++t) acc[t] = v8f{0.f,0.f,0.f,0.f,0.f,0.f,0.f,0.f};

    stage_async16(gsrc, &bsh[0][lofs]);               // slab 0
    for (int s = 0; s < 32; ++s) {
        int k0 = s * 32;
        int buf = s & 1;
        if (s + 1 < 32) {
            stage_async16(gsrc + (s + 1) * 32, &bsh[buf ^ 1][lofs]);
            WAIT_ASYNC_LE1();                         // slab s complete (in-order)
        } else {
            WAIT_ASYNC_LE0();
        }
        __syncthreads();                              // slab s visible to all waves

        __builtin_prefetch(xrow + k0 + 64, 0, 3);     // A stream -> global_prefetch_b8

        ushort16 ta;
#pragma unroll
        for (int i = 0; i < 8; ++i) {
            ta[i]     = xrow[k0 + half * 8 + i];      // K = i + half*8
            ta[8 + i] = xrow[k0 + 16 + half * 8 + i]; // K = 16 + i + half*8
        }
        v16bf a = __builtin_bit_cast(v16bf, ta);

        const unsigned short* bs = bsh[buf];
#pragma unroll
        for (int t = 0; t < 4; ++t) {
            ushort16 tb;
#pragma unroll
            for (int i = 0; i < 16; ++i)
                tb[i] = bs[(t * 16 + m) * 32 + half * 8 + i + (i & 8)];  // K = half*8 + i (+8 for i>=8)
            acc[t] = __builtin_amdgcn_wmma_f32_16x16x32_bf16(false, a, false,
                         __builtin_bit_cast(v16bf, tb), (short)0, acc[t], false, false);
        }
        __syncthreads();                              // all reads done before overwrite
    }

    int row0 = rb * 128 + wid * 16;
#pragma unroll
    for (int t = 0; t < 4; ++t) {
        int n_out = cb * 64 + t * 16 + m;
        float bv = bias[n_out];
        if (mode == 2) {
#pragma unroll
            for (int r = 0; r < 8; ++r) {
                int row = row0 + r + half * 8;
                out_f32[(size_t)row * D_MODEL + n_out] = acc[t][r] + bv;
            }
        } else {
            int h = n_out >> 6, dk = n_out & 63;
#pragma unroll
            for (int r = 0; r < 8; ++r) {
                int row = row0 + r + half * 8;        // = b*SEQ + n
                int b_ = row >> 9, nq = row & 511;
                unsigned short v = to_bf16_u16((acc[t][r] + bv) * scale);
                if (mode == 0)
                    out_bf[(((size_t)b_ * H_HEADS + h) * SEQ + nq) * D_K + dk] = v;
                else
                    out_bf[(((size_t)b_ * H_HEADS + h) * D_K + dk) * SEQ + nq] = v;
            }
        }
    }
}

// ---------------- flash attention, one 16-row q-tile per wave ----------------
// Qbf,Kbf: (b,h,512,64) bf16 (Q pre-scaled by 1/sqrt(dk)); Vt: (b,h,64,512) bf16
// Bias: (b,h,512,512) f32.  Obf: (b*512, 1024) bf16 for the output projection.
__global__ __launch_bounds__(128)
void k_attn(const unsigned short* __restrict__ Qbf,
            const unsigned short* __restrict__ Kbf,
            const unsigned short* __restrict__ Vt,
            const float* __restrict__ Bias,
            unsigned short* __restrict__ Obf) {
    __shared__ unsigned short pls[4][16 * 32];   // per-wave P staging (D->A transpose)
    int wid = threadIdx.x >> 5, lane = threadIdx.x & 31;
    int gw = blockIdx.x * 4 + wid;               // 1024 waves total
    int qt = gw & 31;
    int h  = (gw >> 5) & 15;
    int b  = gw >> 9;
    int m = lane & 15, half = lane >> 4;
    int qbase = qt * 16;

    const unsigned short* Qb = Qbf + ((size_t)b * H_HEADS + h) * SEQ * D_K;
    const unsigned short* Kb = Kbf + ((size_t)b * H_HEADS + h) * SEQ * D_K;
    const unsigned short* Vb = Vt  + ((size_t)b * H_HEADS + h) * D_K * SEQ;
    const float*          Bb = Bias + ((size_t)b * H_HEADS + h) * SEQ * SEQ;

    // Q A-fragments (K=0..31 and K=32..63)
    ushort16 tq0, tq1;
    const unsigned short* qrow = Qb + (size_t)(qbase + m) * D_K;
#pragma unroll
    for (int i = 0; i < 8; ++i) {
        tq0[i]     = qrow[half * 16 + i];
        tq0[8 + i] = qrow[8 + half * 16 + i];
        tq1[i]     = qrow[32 + half * 16 + i];
        tq1[8 + i] = qrow[40 + half * 16 + i];
    }
    v16bf aq0 = __builtin_bit_cast(v16bf, tq0);
    v16bf aq1 = __builtin_bit_cast(v16bf, tq1);

    float mrun[8], lrun[8];
    v8f acc[4];
#pragma unroll
    for (int t = 0; t < 4; ++t) acc[t] = v8f{0.f,0.f,0.f,0.f,0.f,0.f,0.f,0.f};
#pragma unroll
    for (int r = 0; r < 8; ++r) { mrun[r] = -INFINITY; lrun[r] = 0.f; }

    for (int kb = 0; kb < SEQ; kb += 32) {
        // ---- S = Q @ K^T tiles (16x16 each, cols kb..+15 and kb+16..+31) ----
        v8f s0 = v8f{0.f,0.f,0.f,0.f,0.f,0.f,0.f,0.f};
        v8f s1 = s0;
        const unsigned short* kr0 = Kb + (size_t)(kb + m) * D_K;
        const unsigned short* kr1 = Kb + (size_t)(kb + 16 + m) * D_K;
        ushort16 tk;
#pragma unroll
        for (int i = 0; i < 16; ++i) tk[i] = kr0[half * 16 + i];
        s0 = __builtin_amdgcn_wmma_f32_16x16x32_bf16(false, aq0, false,
                 __builtin_bit_cast(v16bf, tk), (short)0, s0, false, false);
#pragma unroll
        for (int i = 0; i < 16; ++i) tk[i] = kr0[32 + half * 16 + i];
        s0 = __builtin_amdgcn_wmma_f32_16x16x32_bf16(false, aq1, false,
                 __builtin_bit_cast(v16bf, tk), (short)0, s0, false, false);
#pragma unroll
        for (int i = 0; i < 16; ++i) tk[i] = kr1[half * 16 + i];
        s1 = __builtin_amdgcn_wmma_f32_16x16x32_bf16(false, aq0, false,
                 __builtin_bit_cast(v16bf, tk), (short)0, s1, false, false);
#pragma unroll
        for (int i = 0; i < 16; ++i) tk[i] = kr1[32 + half * 16 + i];
        s1 = __builtin_amdgcn_wmma_f32_16x16x32_bf16(false, aq1, false,
                 __builtin_bit_cast(v16bf, tk), (short)0, s1, false, false);

        // ---- add precomputed positional bias ----
#pragma unroll
        for (int r = 0; r < 8; ++r) {
            const float* brow = Bb + (size_t)(qbase + r + half * 8) * SEQ + kb;
            s0[r] += brow[m];
            s1[r] += brow[16 + m];
        }

        // ---- online softmax (rows live across the 16 lanes of each half) ----
        float corr[8];
#pragma unroll
        for (int r = 0; r < 8; ++r) {
            float mx = fmaxf(s0[r], s1[r]);
#pragma unroll
            for (int d = 1; d < 16; d <<= 1) mx = fmaxf(mx, __shfl_xor(mx, d, 32));
            float mn = fmaxf(mrun[r], mx);
            corr[r] = __expf(mrun[r] - mn);
            mrun[r] = mn;
            float p0 = __expf(s0[r] - mn);
            float p1 = __expf(s1[r] - mn);
            s0[r] = p0; s1[r] = p1;
            float sum = p0 + p1;
#pragma unroll
            for (int d = 1; d < 16; d <<= 1) sum += __shfl_xor(sum, d, 32);
            lrun[r] = lrun[r] * corr[r] + sum;
        }
#pragma unroll
        for (int t = 0; t < 4; ++t)
#pragma unroll
            for (int r = 0; r < 8; ++r) acc[t][r] *= corr[r];

        // ---- D-layout P -> LDS -> A-layout P (per-wave, DS ops are in-order) ----
        unsigned short* pw = pls[wid];
#pragma unroll
        for (int r = 0; r < 8; ++r) {
            int row = r + half * 8;
            pw[row * 32 + m]      = to_bf16_u16(s0[r]);
            pw[row * 32 + 16 + m] = to_bf16_u16(s1[r]);
        }
        asm volatile("s_wait_dscnt 0" ::: "memory");
        ushort16 tp;
        const unsigned short* pr = pw + m * 32;
#pragma unroll
        for (int i = 0; i < 8; ++i) {
            tp[i]     = pr[half * 16 + i];
            tp[8 + i] = pr[8 + half * 16 + i];
        }
        v16bf pa = __builtin_bit_cast(v16bf, tp);

        // ---- O += P @ V (V stored d-major -> contiguous B fragments) ----
#pragma unroll
        for (int t = 0; t < 4; ++t) {
            const unsigned short* vrow = Vb + (size_t)(t * 16 + m) * SEQ + kb;
            ushort16 tv;
#pragma unroll
            for (int i = 0; i < 16; ++i) tv[i] = vrow[half * 16 + i];
            acc[t] = __builtin_amdgcn_wmma_f32_16x16x32_bf16(false, pa, false,
                         __builtin_bit_cast(v16bf, tv), (short)0, acc[t], false, false);
        }
    }

    // ---- epilogue: divide by l, store bf16 (b*512+q, h*64+d) for O-projection ----
#pragma unroll
    for (int t = 0; t < 4; ++t)
#pragma unroll
        for (int r = 0; r < 8; ++r) {
            int qr = qbase + r + half * 8;
            float o = acc[t][r] / lrun[r];
            Obf[((size_t)b * SEQ + qr) * D_MODEL + h * D_K + t * 16 + m] = to_bf16_u16(o);
        }
}

// ---------------- host launcher ----------------
extern "C" void kernel_launch(void* const* d_in, const int* in_sizes, int n_in,
                              void* d_out, int out_size, void* d_ws, size_t ws_size,
                              hipStream_t stream) {
    (void)in_sizes; (void)n_in; (void)out_size; (void)ws_size;
    const float* query   = (const float*)d_in[0];
    const float* key     = (const float*)d_in[1];
    const float* value   = (const float*)d_in[2];
    const float* query_x = (const float*)d_in[3];
    const float* query_y = (const float*)d_in[4];
    const float* key_x   = (const float*)d_in[5];
    const float* key_y   = (const float*)d_in[6];
    const float* Wq = (const float*)d_in[7];   const float* bq = (const float*)d_in[8];
    const float* Wk = (const float*)d_in[9];   const float* bk = (const float*)d_in[10];
    const float* Wv = (const float*)d_in[11];  const float* bv = (const float*)d_in[12];
    const float* Wo = (const float*)d_in[13];  const float* bo = (const float*)d_in[14];
    const float* x_table = (const float*)d_in[15];
    const float* y_table = (const float*)d_in[16];
    const float* Wb = (const float*)d_in[17];
    const float* bb = (const float*)d_in[18];

    char* ws = (char*)d_ws;
    size_t off = 0;
    auto alloc = [&](size_t bytes) -> void* {
        void* p = ws + off; off += (bytes + 255) & ~(size_t)255; return p;
    };
    const size_t NN = (size_t)D_MODEL * D_MODEL;       // 1024*1024
    unsigned short* xq  = (unsigned short*)alloc(NN * 2);
    unsigned short* xk  = (unsigned short*)alloc(NN * 2);
    unsigned short* xv  = (unsigned short*)alloc(NN * 2);
    unsigned short* wqT = (unsigned short*)alloc(NN * 2);
    unsigned short* wkT = (unsigned short*)alloc(NN * 2);
    unsigned short* wvT = (unsigned short*)alloc(NN * 2);
    unsigned short* woT = (unsigned short*)alloc(NN * 2);
    unsigned short* qbf = (unsigned short*)alloc(NN * 2);
    unsigned short* kbf = (unsigned short*)alloc(NN * 2);
    unsigned short* vtb = (unsigned short*)alloc(NN * 2);
    unsigned short* obf = (unsigned short*)alloc(NN * 2);
    float* px   = (float*)alloc((size_t)N_EMB * 16 * 4);
    float* py   = (float*)alloc((size_t)N_EMB * 16 * 4);
    float* bias = (float*)alloc((size_t)BATCH * H_HEADS * SEQ * SEQ * 4);

    int n = (int)NN;
    k_f32_to_bf16<<<(n + 255) / 256, 256, 0, stream>>>(query, xq, n);
    k_f32_to_bf16<<<(n + 255) / 256, 256, 0, stream>>>(key,   xk, n);
    k_f32_to_bf16<<<(n + 255) / 256, 256, 0, stream>>>(value, xv, n);
    dim3 tg(D_MODEL / 256, D_MODEL);
    k_transpose_bf16<<<tg, 256, 0, stream>>>(Wq, wqT, D_MODEL, D_MODEL);
    k_transpose_bf16<<<tg, 256, 0, stream>>>(Wk, wkT, D_MODEL, D_MODEL);
    k_transpose_bf16<<<tg, 256, 0, stream>>>(Wv, wvT, D_MODEL, D_MODEL);
    k_transpose_bf16<<<tg, 256, 0, stream>>>(Wo, woT, D_MODEL, D_MODEL);

    int tp = N_EMB * 16;
    k_table_proj<<<(tp + 255) / 256, 256, 0, stream>>>(x_table, Wb, px, N_EMB);
    k_table_proj<<<(tp + 255) / 256, 256, 0, stream>>>(y_table, Wb, py, N_EMB);
    k_bias<<<dim3(SEQ / 256, SEQ, BATCH), 256, 0, stream>>>(
        query_x, query_y, key_x, key_y, px, py, bb, bias);

    // Q scaled by 1/sqrt(d_k) at store time; K plain; V transposed per head.
    k_gemm<<<128, 256, 0, stream>>>(xq, wqT, bq, 0.125f, 0, qbf, nullptr);
    k_gemm<<<128, 256, 0, stream>>>(xk, wkT, bk, 1.0f,   0, kbf, nullptr);
    k_gemm<<<128, 256, 0, stream>>>(xv, wvT, bv, 1.0f,   1, vtb, nullptr);

    k_attn<<<256, 128, 0, stream>>>(qbf, kbf, vtb, bias, obf);

    k_gemm<<<128, 256, 0, stream>>>(obf, woT, bo, 1.0f, 2, nullptr, (float*)d_out);
}
